// SRNN_ALIF_71382356459525
// MI455X (gfx1250) — matrix operand order repaired and checked by reading, not compiled, not run on hardware
//
#include <hip/hip_runtime.h>
#include <hip/hip_bf16.h>
#include <math.h>

// Problem sizes (fixed by the reference)
#define B_    128
#define T_    500
#define I_    700
#define H_    256
#define O_    20
#define KPAD  704          // 22 * 32 (K of the big GEMM padded to WMMA chunk)
#define KC_I  22
#define BJ0   0.01f
#define SPK_STRIDE 264     // 256 + 8 bf16 pad -> conflict-free b128 LDS reads

typedef __bf16 bf16_t;
typedef __bf16 v16bf __attribute__((ext_vector_type(16)));
typedef float  v8f   __attribute__((ext_vector_type(8)));

__device__ __forceinline__ v8f wmma_bf16(v16bf a, v16bf b, v8f c) {
  // D = A(16x32 bf16) x B(32x16 bf16) + C(16x16 f32)
  return __builtin_amdgcn_wmma_f32_16x16x32_bf16(false, a, false, b, (short)0, c,
                                                 false, false);
}

__device__ __forceinline__ void split_bf16(float x, bf16_t& hi, bf16_t& lo) {
  hi = (bf16_t)x;
  lo = (bf16_t)(x - (float)hi);
}

// Build a 16-element bf16 fragment from two 16B-aligned 8-element runs
__device__ __forceinline__ v16bf frag2(const bf16_t* p0, const bf16_t* p1) {
  union { uint4 u[2]; v16bf v; } f;
  f.u[0] = *(const uint4*)p0;
  f.u[1] = *(const uint4*)p1;
  return f.v;
}
// Build a fragment from one 32B-aligned run of 16 bf16
__device__ __forceinline__ v16bf frag1(const bf16_t* p) {
  union { uint4 u[2]; v16bf v; } f;
  const uint4* q = (const uint4*)p;
  f.u[0] = q[0];
  f.u[1] = q[1];
  return f.v;
}

// ---------------------------------------------------------------------------
// Async global->LDS copy (ASYNCcnt path), with synchronous fallback.
// Builtin signature (from hipcc diagnostic): param 1 is a non-const
// int-vector(16B) pointer in the global (AS1) address space.
// ---------------------------------------------------------------------------
#if defined(__gfx1250__) && __has_builtin(__builtin_amdgcn_global_load_async_to_lds_b128)
#define HAVE_ASYNC_LDS 1
#else
#define HAVE_ASYNC_LDS 0
#endif

typedef int vi4_t __attribute__((vector_size(16)));
typedef vi4_t __attribute__((address_space(1)))* gvi4_p;   // global
typedef vi4_t __attribute__((address_space(3)))* lvi4_p;   // LDS

__device__ __forceinline__ void async_g2l_b128(const float* g, float* l) {
#if HAVE_ASYNC_LDS
  __builtin_amdgcn_global_load_async_to_lds_b128(
      (gvi4_p)(void*)const_cast<float*>(g), (lvi4_p)(void*)l, 0, 0);
#else
  *(uint4*)l = *(const uint4*)g;   // synchronous fallback (still correct)
#endif
}

template <int N>
__device__ __forceinline__ void wait_async() {
#if HAVE_ASYNC_LDS
#if __has_builtin(__builtin_amdgcn_s_wait_asynccnt)
  __builtin_amdgcn_s_wait_asynccnt((unsigned short)N);
#else
  asm volatile("s_wait_asynccnt %0" ::"n"(N) : "memory");
#endif
#endif
}

// ---------------------------------------------------------------------------
// Kernel 1a: pre-swizzle W_i2h into WMMA B-fragment layout, bf16 hi/lo split.
// B[k][n] = W_i2h[n][k]; swizzled index = ((n*KC_I + kc)*2 + half)*16 + i,
// kc=k/32, half=(k%32)/16, i=k%16. K padded to 704 with zeros.
// ---------------------------------------------------------------------------
__global__ __launch_bounds__(256)
void swizzle_wi2h_kernel(const float* __restrict__ W,
                         bf16_t* __restrict__ hiB, bf16_t* __restrict__ loB) {
  int gid = blockIdx.x * 256 + threadIdx.x;      // grid = KPAD blocks
  int n = gid / KPAD;
  int k = gid - n * KPAD;
  float v = (k < I_) ? W[(size_t)n * I_ + k] : 0.0f;
  bf16_t h, l;
  split_bf16(v, h, l);
  int kc = k >> 5, r = k & 31, half = r >> 4, i = r & 15;
  int idx = ((n * KC_I + kc) * 2 + half) * 16 + i;
  hiB[idx] = h;
  loB[idx] = l;
}

// ---------------------------------------------------------------------------
// Kernel 1b: pre-swizzle W_h2o (N padded 20->32, K=256), bf16 hi/lo split.
// ---------------------------------------------------------------------------
__global__ __launch_bounds__(256)
void swizzle_who_kernel(const float* __restrict__ W,
                        bf16_t* __restrict__ hiB, bf16_t* __restrict__ loB) {
  int gid = blockIdx.x * 256 + threadIdx.x;      // 32 blocks -> 8192
  int n = gid >> 8, k = gid & 255;
  float v = (n < O_) ? W[(size_t)n * H_ + k] : 0.0f;
  bf16_t h, l;
  split_bf16(v, h, l);
  int idx = ((n * 8 + (k >> 5)) * 2 + ((k & 31) >> 4)) * 16 + (k & 15);
  hiB[idx] = h;
  loB[idx] = l;
}

// ---------------------------------------------------------------------------
// Kernel 2: xw[t][b][h] = x[b,t,:] @ W_i2h.T + b_i2h + b_h2h
// Block (8 waves) computes 16(batch) x 256(H) for one t. Software-pipelined:
// the 16x32 A chunk for kc+1 is prefetched into registers while kc computes,
// so global latency hides behind the 6 WMMAs per chunk. 3-term hi/lo split
// (a_hi*b_hi + a_hi*b_lo + a_lo*b_hi) gives ~fp32 accuracy.
// ---------------------------------------------------------------------------
__global__ __launch_bounds__(256)
void gemm_xw_kernel(const float* __restrict__ x,
                    const bf16_t* __restrict__ whi, const bf16_t* __restrict__ wlo,
                    const float* __restrict__ b_i2h, const float* __restrict__ b_h2h,
                    float* __restrict__ xw) {
  __shared__ __align__(16) bf16_t Ahi[16][40];   // stride 40 -> conflict-free
  __shared__ __align__(16) bf16_t Alo[16][40];

  int blk = blockIdx.x;
  int b16 = blk / T_;
  int t   = blk - b16 * T_;
  int b0  = b16 * 16;

  int tid  = threadIdx.x;
  int lane = tid & 31;
  int w    = tid >> 5;
  int half = lane >> 4;
  int l15  = lane & 15;

  int n0 = w * 32 + l15;      // wave covers two 16-wide N tiles
  int n1 = n0 + 16;
  float bias0 = b_i2h[n0] + b_h2h[n0];
  float bias1 = b_i2h[n1] + b_h2h[n1];

  // staging assignment: thread -> two (row, kk) slots of the 16x32 chunk
  int rowA[2], kkA[2];
#pragma unroll
  for (int j = 0; j < 2; ++j) {
    int e = tid + j * 256;
    rowA[j] = e >> 5;
    kkA[j]  = e & 31;
  }

  // prologue: load chunk kc=0 into registers
  float areg[2];
#pragma unroll
  for (int j = 0; j < 2; ++j) {
    int k = kkA[j];
    areg[j] = (k < I_) ? x[(size_t)((b0 + rowA[j]) * T_ + t) * I_ + k] : 0.0f;
  }

  v8f acc0 = {};
  v8f acc1 = {};

  for (int kc = 0; kc < KC_I; ++kc) {
    // convert & publish the chunk loaded last iteration
#pragma unroll
    for (int j = 0; j < 2; ++j) {
      bf16_t h, l;
      split_bf16(areg[j], h, l);
      Ahi[rowA[j]][kkA[j]] = h;
      Alo[rowA[j]][kkA[j]] = l;
    }
    __syncthreads();

    // prefetch chunk kc+1 (consumed next iteration; latency hidden by WMMAs)
    if (kc + 1 < KC_I) {
#pragma unroll
      for (int j = 0; j < 2; ++j) {
        int k = (kc + 1) * 32 + kkA[j];
        areg[j] = (k < I_) ? x[(size_t)((b0 + rowA[j]) * T_ + t) * I_ + k] : 0.0f;
      }
    }

    // A fragments (ds b128 x2 each)
    v16bf fa_hi = frag2(&Ahi[l15][half * 16], &Ahi[l15][half * 16 + 8]);
    v16bf fa_lo = frag2(&Alo[l15][half * 16], &Alo[l15][half * 16 + 8]);

    // B fragments from the pre-swizzled L2-resident buffer
    int bi0 = ((n0 * KC_I + kc) * 2 + half) * 16;
    int bi1 = ((n1 * KC_I + kc) * 2 + half) * 16;
    v16bf fb0h = frag1(whi + bi0);
    v16bf fb0l = frag1(wlo + bi0);
    v16bf fb1h = frag1(whi + bi1);
    v16bf fb1l = frag1(wlo + bi1);

    acc0 = wmma_bf16(fa_hi, fb0h, acc0);
    acc0 = wmma_bf16(fa_hi, fb0l, acc0);
    acc0 = wmma_bf16(fa_lo, fb0h, acc0);
    acc1 = wmma_bf16(fa_hi, fb1h, acc1);
    acc1 = wmma_bf16(fa_hi, fb1l, acc1);
    acc1 = wmma_bf16(fa_lo, fb1h, acc1);
    __syncthreads();
  }

  // C layout: element r -> M = r + half*8, N = n-tile base + (lane&15)
#pragma unroll
  for (int r = 0; r < 8; ++r) {
    int m = r + half * 8;
    size_t o = (size_t)(t * B_ + b0 + m) * H_;
    xw[o + n0] = acc0[r] + bias0;
    xw[o + n1] = acc1[r] + bias1;
  }
}

// ---------------------------------------------------------------------------
// Kernel 3: sequential ALIF scan. 8 workgroups x 16 batch rows.
// W_h2h hi+lo (256KB) stays resident in LDS; xw slabs are double-buffered in
// LDS via GLOBAL_LOAD_ASYNC_TO_LDS_B128 (per-wave ASYNCcnt, wave-private
// regions -> no extra barriers). W_h2o hi/lo streams from L2.
// ---------------------------------------------------------------------------
#define SMEM_BF16_ELEMS (2 * H_ * H_ + 16 * SPK_STRIDE)
#define SMEM_F32_ELEMS  (2 * 16 * H_ + 16 * 32 + 3 * 16 * O_ + 64)
#define SCAN_SMEM_BYTES (SMEM_BF16_ELEMS * 2 + SMEM_F32_ELEMS * 4)   // ~302KB

__global__ __launch_bounds__(256)
void scan_kernel(const float* __restrict__ xw,
                 const float* __restrict__ Whh,
                 const bf16_t* __restrict__ who_hi_g,
                 const bf16_t* __restrict__ who_lo_g,
                 const float* __restrict__ b_h2o,
                 const float* __restrict__ tau_adp_o,
                 const float* __restrict__ tau_m_h,
                 const float* __restrict__ tau_m_o,
                 const float* __restrict__ h0,
                 const float* __restrict__ o0,
                 float* __restrict__ out) {
  extern __shared__ char smem[];
  bf16_t* whh_hi = (bf16_t*)smem;                    // 65536 bf16
  bf16_t* whh_lo = whh_hi + H_ * H_;                 // 65536
  bf16_t* spk    = whh_lo + H_ * H_;                 // 16 x 264
  float*  xwbuf  = (float*)(spk + 16 * SPK_STRIDE);  // 2 x 16 x 256 (dbl buf)
  float*  oin    = xwbuf + 2 * 16 * H_;              // 16 x 32
  float*  omem   = oin + 16 * 32;                    // 16 x 20
  float*  ospk   = omem + 16 * O_;
  float*  bo     = ospk + 16 * O_;
  float*  oco    = bo + 16 * O_;   // [0..19] bias, [20..39] alpha_o, [40..59] ro_o

  int tid  = threadIdx.x;
  int lane = tid & 31;
  int w    = tid >> 5;
  int half = lane >> 4;
  int l15  = lane & 15;
  int b0   = blockIdx.x * 16;

  // --- init: swizzle W_h2h into LDS (hi/lo) ---
  for (int e = tid; e < H_ * H_; e += 256) {
    int n = e >> 8, k = e & 255;
    bf16_t h, l;
    split_bf16(Whh[(size_t)n * H_ + k], h, l);
    int idx = ((n * 8 + (k >> 5)) * 2 + ((k & 31) >> 4)) * 16 + (k & 15);
    whh_hi[idx] = h;
    whh_lo[idx] = l;
  }
  if (tid < O_) {
    oco[tid]      = b_h2o[tid];
    oco[20 + tid] = expf(-1.0f / tau_m_o[tid]);
    oco[40 + tid] = expf(-1.0f / tau_adp_o[tid]);
  }
  for (int e = tid; e < 16 * O_; e += 256) {
    int b = e / O_, o = e - b * O_;
    float v = o0[(b0 + b) * O_ + o];
    omem[e] = v;       // carry0: o_mem = o0
    ospk[e] = v;       // carry0: o_spk = o0
    bo[e]   = BJ0;     // carry0: b_o = b_j0
  }

  // Hidden state in registers, aligned to C-fragment layout.
  float hm[2][8], sp[2][8], alph[2];
  int nn0 = w * 32 + l15;
  int nn1 = nn0 + 16;
  alph[0] = expf(-1.0f / tau_m_h[nn0]);
  alph[1] = expf(-1.0f / tau_m_h[nn1]);
#pragma unroll
  for (int j = 0; j < 2; ++j) {
    int n = j ? nn1 : nn0;
#pragma unroll
    for (int r = 0; r < 8; ++r) {
      int m = r + half * 8;
      float v = h0[(b0 + m) * H_ + n];
      hm[j][r] = v;                     // carry0: h_mem = h0
      sp[j][r] = v;                     // carry0: h_spk = h0
      spk[m * SPK_STRIDE + n] = (bf16_t)v;
    }
  }

  // async-staging assignment: wave w owns cols [32w, 32w+32) of the slab
  int arow = lane >> 3;                 // 0..3
  int acol = w * 32 + (lane & 7) * 4;   // 16B granule
  // prologue: pull xw slab for t=0 into buffer 0
  {
    const float* xg = xw + (size_t)b0 * H_;
#pragma unroll
    for (int q = 0; q < 4; ++q) {
      int r = q * 4 + arow;
      async_g2l_b128(xg + (size_t)r * H_ + acol, xwbuf + r * H_ + acol);
    }
  }
  __syncthreads();

  // --- sequential scan over T ---
  for (int t = 0; t < T_; ++t) {
    // stage next step's xw slab into the other buffer (wave-private region)
    if (t + 1 < T_) {
      const float* xg = xw + (size_t)(t + 1) * (B_ * H_) + (size_t)b0 * H_;
      float* xl = xwbuf + ((t + 1) & 1) * (16 * H_);
#pragma unroll
      for (int q = 0; q < 4; ++q) {
        int r = q * 4 + arow;
        async_g2l_b128(xg + (size_t)r * H_ + acol, xl + r * H_ + acol);
      }
    }

    // recurrent GEMM: spk(prev) [16x256] x W_h2h.T [256x256], W split hi/lo
    v8f acc0 = {};
    v8f acc1 = {};
#pragma unroll
    for (int kc = 0; kc < 8; ++kc) {
      v16bf fa = frag2(spk + l15 * SPK_STRIDE + kc * 32 + half * 16,
                       spk + l15 * SPK_STRIDE + kc * 32 + half * 16 + 8);
      int bi0 = ((nn0 * 8 + kc) * 2 + half) * 16;
      int bi1 = ((nn1 * 8 + kc) * 2 + half) * 16;
      acc0 = wmma_bf16(fa, frag1(whh_hi + bi0), acc0);
      acc0 = wmma_bf16(fa, frag1(whh_lo + bi0), acc0);
      acc1 = wmma_bf16(fa, frag1(whh_hi + bi1), acc1);
      acc1 = wmma_bf16(fa, frag1(whh_lo + bi1), acc1);
    }
    __syncthreads();   // (a) all waves done reading spk(prev)

    // wait for this step's async slab (4 newest outstanding are for t+1)
    if (t + 1 < T_) wait_async<4>();
    else            wait_async<0>();

    // ALIF hidden update; write new spikes
    const float* xb = xwbuf + (t & 1) * (16 * H_);
#pragma unroll
    for (int j = 0; j < 2; ++j) {
      int n = j ? nn1 : nn0;
      float a = alph[j];
#pragma unroll
      for (int r = 0; r < 8; ++r) {
        float hin = xb[(r + half * 8) * H_ + n] + (j ? acc1[r] : acc0[r]);
        float h   = hm[j][r] * a + (1.0f - a) * hin - BJ0 * sp[j][r];
        float ns  = (h - BJ0) > 0.0f ? 1.0f : 0.0f;
        hm[j][r] = h;
        sp[j][r] = ns;
        spk[(r + half * 8) * SPK_STRIDE + n] = (bf16_t)ns;  // exact in bf16
      }
    }
    __syncthreads();   // (b) new spikes visible

    // output GEMM: spk(new) [16x256] x W_h2o.T [256x32-padded], B from L2
    if (w < 2) {
      int n = w * 16 + l15;
      v8f oacc = {};
#pragma unroll
      for (int kc = 0; kc < 8; ++kc) {
        v16bf fa = frag2(spk + l15 * SPK_STRIDE + kc * 32 + half * 16,
                         spk + l15 * SPK_STRIDE + kc * 32 + half * 16 + 8);
        int bi = ((n * 8 + kc) * 2 + half) * 16;
        oacc = wmma_bf16(fa, frag1(who_hi_g + bi), oacc);
        oacc = wmma_bf16(fa, frag1(who_lo_g + bi), oacc);
      }
#pragma unroll
      for (int r = 0; r < 8; ++r)
        oin[(r + half * 8) * 32 + n] = oacc[r];
    }
    __syncthreads();   // (c) o_in visible

    // output-layer ALIF update + log_softmax; one lane per batch row
    if (tid < 16) {
      int b = tid;
      float om[O_];
      float mx = -3.0e38f;
#pragma unroll
      for (int o = 0; o < O_; ++o) {
        float oi  = oin[b * 32 + o] + oco[o];
        float ro  = oco[40 + o];
        float ao  = oco[20 + o];
        float bov = bo[b * O_ + o];
        float osp = ospk[b * O_ + o];
        float m   = omem[b * O_ + o];
        bov = ro * bov + (1.0f - ro) * osp;           // adaptation (isAdapt=1)
        float Bo = BJ0 + 1.8f * bov;
        m = m * ao + (1.0f - ao) * oi - Bo * osp;
        float ns = (m - Bo) > 0.0f ? 1.0f : 0.0f;
        omem[b * O_ + o] = m;
        ospk[b * O_ + o] = ns;
        bo[b * O_ + o]   = bov;
        om[o] = m;
        mx = fmaxf(mx, m);
      }
      float se = 0.0f;
#pragma unroll
      for (int o = 0; o < O_; ++o) se += expf(om[o] - mx);
      float lse = mx + logf(se);
#pragma unroll
      for (int o = 0; o < O_; ++o)
        out[((size_t)(b0 + b) * O_ + o) * T_ + t] = om[o] - lse;   // [B,O,T]
    }
  }
}

// ---------------------------------------------------------------------------
// Host-side launcher
// ---------------------------------------------------------------------------
#define XW_BYTES   ((size_t)T_ * B_ * H_ * 4)      // 65,536,000
#define WSW_BYTES  ((size_t)H_ * KPAD * 2)         // 360,448 per hi/lo buffer
#define WHO_BYTES  ((size_t)32 * H_ * 2)           // 16,384 per hi/lo buffer

extern "C" void kernel_launch(void* const* d_in, const int* in_sizes, int n_in,
                              void* d_out, int out_size, void* d_ws, size_t ws_size,
                              hipStream_t stream) {
  (void)in_sizes; (void)n_in; (void)out_size; (void)ws_size;

  const float* x         = (const float*)d_in[0];
  const float* W_i2h     = (const float*)d_in[1];
  const float* b_i2h     = (const float*)d_in[2];
  const float* W_h2h     = (const float*)d_in[3];
  const float* b_h2h     = (const float*)d_in[4];
  const float* W_h2o     = (const float*)d_in[5];
  const float* b_h2o     = (const float*)d_in[6];
  // d_in[7] = tau_adp_h: dead in the reference (hidden adaptation disabled)
  const float* tau_adp_o = (const float*)d_in[8];
  const float* tau_m_h   = (const float*)d_in[9];
  const float* tau_m_o   = (const float*)d_in[10];
  const float* h0        = (const float*)d_in[11];
  const float* o0        = (const float*)d_in[12];
  float* out = (float*)d_out;

  char* ws = (char*)d_ws;
  float*  xw     = (float*)ws;
  bf16_t* whi    = (bf16_t*)(ws + XW_BYTES);
  bf16_t* wlo    = (bf16_t*)(ws + XW_BYTES + WSW_BYTES);
  bf16_t* who_hi = (bf16_t*)(ws + XW_BYTES + 2 * WSW_BYTES);
  bf16_t* who_lo = (bf16_t*)(ws + XW_BYTES + 2 * WSW_BYTES + WHO_BYTES);

  // 1) pre-swizzle weights (hi/lo) into WMMA B layout
  swizzle_wi2h_kernel<<<KPAD, 256, 0, stream>>>(W_i2h, whi, wlo);
  swizzle_who_kernel<<<32, 256, 0, stream>>>(W_h2o, who_hi, who_lo);

  // 2) big GEMM: xw[t][b][h] = x@W_i2h.T + b_i2h + b_h2h (software-pipelined)
  gemm_xw_kernel<<<(B_ / 16) * T_, 256, 0, stream>>>(x, whi, wlo, b_i2h, b_h2h, xw);

  // 3) sequential ALIF scan (8 WGs, ~302KB LDS, async xw double-buffering)
  (void)hipFuncSetAttribute(reinterpret_cast<const void*>(scan_kernel),
                            hipFuncAttributeMaxDynamicSharedMemorySize,
                            (int)SCAN_SMEM_BYTES);
  scan_kernel<<<B_ / 16, 256, SCAN_SMEM_BYTES, stream>>>(
      xw, W_h2h, who_hi, who_lo, b_h2o, tau_adp_o, tau_m_h, tau_m_o, h0, o0, out);
}
